// MultiSLNet_14723147890778
// MI455X (gfx1250) — matrix operand order repaired and checked by reading, not compiled, not run on hardware
//
#include <hip/hip_runtime.h>
#include <hip/hip_bf16.h>

typedef __attribute__((ext_vector_type(16))) _Float16 v16h;
typedef __attribute__((ext_vector_type(8)))  _Float16 v8h;
typedef __attribute__((ext_vector_type(8)))  float    v8f;

#define B_N        32768
#define NMODELS    64
#define IN_F       6
#define OUT_F      3
#define HID        64
#define MAX_TILES  (B_N / 16 + NMODELS)          // 2112 padded expert tiles
#define MAX_WGS    (B_N / 64 + NMODELS)          // 576 workgroups (4 tiles each)
#define NWAVES     2

// Wave-local LDS ordering: all lanes of the wave issue this together, so once
// DScnt==0 every lane's LDS writes/reads are complete (CDNA5 split counters).
__device__ __forceinline__ void wave_lds_fence() {
    asm volatile("s_wait_dscnt 0x0" ::: "memory");
}

// ---------------------------------------------------------------------------
// Fragment builders (wave32 WMMA 16x16x32 f16 layouts):
// A (16x32): lane m=L%16,g=L/16; e -> K = (e<8?e:e+8)+8g  => two contiguous v8h
// B (32x16): lane n=L%16,g=L/16; e -> K = (e<8?e:e+8)+8g  => two contiguous v8h
//            when W is stored transposed Wt[n][k]
// C (16x16 f32): VGPR r, lane L -> M=r+8g, N=L%16
// ---------------------------------------------------------------------------
__device__ __forceinline__ v16h frag_a(const _Float16* tile, int kbase, int lane) {
    const int m = lane & 15, g = lane >> 4;
    const v8h lo = *(const v8h*)(tile + m * 64 + kbase + 8 * g);
    const v8h hi = *(const v8h*)(tile + m * 64 + kbase + 16 + 8 * g);
    return __builtin_shufflevector(lo, hi, 0, 1, 2, 3, 4, 5, 6, 7,
                                   8, 9, 10, 11, 12, 13, 14, 15);
}

__device__ __forceinline__ v16h frag_b(const _Float16* Wt, int ldk, int ncol0,
                                       int kbase, int lane) {
    const int n = lane & 15, g = lane >> 4;
    const v8h lo = *(const v8h*)(Wt + (ncol0 + n) * ldk + kbase + 8 * g);
    const v8h hi = *(const v8h*)(Wt + (ncol0 + n) * ldk + kbase + 16 + 8 * g);
    return __builtin_shufflevector(lo, hi, 0, 1, 2, 3, 4, 5, 6, 7,
                                   8, 9, 10, 11, 12, 13, 14, 15);
}

// 16x64 @ 64x64 + bias, ReLU: src(LDS f16) -> dst(LDS f16). 8 WMMAs.
__device__ __forceinline__ void dense64(const _Float16* src, _Float16* dst,
                                        const _Float16* Wt, const float* bias, int lane) {
    const int n = lane & 15, g = lane >> 4;
    const v16h a0 = frag_a(src, 0, lane);
    const v16h a1 = frag_a(src, 32, lane);
    v8f acc[4];
#pragma unroll
    for (int nt = 0; nt < 4; ++nt) {
        v8f c;
        const float bv = bias[nt * 16 + n];
#pragma unroll
        for (int r = 0; r < 8; ++r) c[r] = bv;
        c = __builtin_amdgcn_wmma_f32_16x16x32_f16(false, a0, false,
                frag_b(Wt, 64, nt * 16, 0, lane), (short)0, c, false, false);
        c = __builtin_amdgcn_wmma_f32_16x16x32_f16(false, a1, false,
                frag_b(Wt, 64, nt * 16, 32, lane), (short)0, c, false, false);
        acc[nt] = c;
    }
#pragma unroll
    for (int nt = 0; nt < 4; ++nt)
#pragma unroll
        for (int r = 0; r < 8; ++r) {
            float v = acc[nt][r];
            v = v > 0.0f ? v : 0.0f;
            dst[(r + 8 * g) * 64 + nt * 16 + n] = (_Float16)v;
        }
}

// Input layer: 16x32 (x padded) @ 32x64 + bias, ReLU. 4 WMMAs.
__device__ __forceinline__ void dense_in(const _Float16* src, _Float16* dst,
                                         const _Float16* Wt0, const float* bias, int lane) {
    const int n = lane & 15, g = lane >> 4;
    const v16h a = frag_a(src, 0, lane);
    v8f acc[4];
#pragma unroll
    for (int nt = 0; nt < 4; ++nt) {
        v8f c;
        const float bv = bias[nt * 16 + n];
#pragma unroll
        for (int r = 0; r < 8; ++r) c[r] = bv;
        c = __builtin_amdgcn_wmma_f32_16x16x32_f16(false, a, false,
                frag_b(Wt0, 32, nt * 16, 0, lane), (short)0, c, false, false);
        acc[nt] = c;
    }
#pragma unroll
    for (int nt = 0; nt < 4; ++nt)
#pragma unroll
        for (int r = 0; r < 8; ++r) {
            float v = acc[nt][r];
            v = v > 0.0f ? v : 0.0f;
            dst[(r + 8 * g) * 64 + nt * 16 + n] = (_Float16)v;
        }
}

// Head: 16x64 @ 64x16 (cols 3..15 zero-padded at staging). 2 WMMAs.
__device__ __forceinline__ v8f head(const _Float16* src, const _Float16* Wt4,
                                    const float* b4pad, int lane) {
    const int n = lane & 15;
    v8f c;
    const float bv = b4pad[n];
#pragma unroll
    for (int r = 0; r < 8; ++r) c[r] = bv;
    c = __builtin_amdgcn_wmma_f32_16x16x32_f16(false, frag_a(src, 0, lane), false,
            frag_b(Wt4, 64, 0, 0, lane), (short)0, c, false, false);
    c = __builtin_amdgcn_wmma_f32_16x16x32_f16(false, frag_a(src, 32, lane), false,
            frag_b(Wt4, 64, 0, 32, lane), (short)0, c, false, false);
    return c;
}

__device__ __forceinline__ void store_out(float* out, const int* rows, int cnt,
                                          int lod, v8f c, int lane) {
    const int n = lane & 15, g = lane >> 4;
#pragma unroll
    for (int r = 0; r < 8; ++r) {
        const int mm = r + 8 * g;
        if (n < OUT_F && mm < cnt)
            out[rows[mm] * (4 * OUT_F) + lod * OUT_F + n] = c[r];
    }
}

// Stage 64x64 f32 W[k][n] -> transposed LDS f16 Wt[n][k]. Coalesced reads.
__device__ __forceinline__ void stage_w64(const float* __restrict__ W,
                                          _Float16* Wt, int tid) {
    for (int idx = tid; idx < 64 * 64; idx += 64) {
        const int k = idx >> 6, n = idx & 63;
        Wt[(n << 6) + k] = (_Float16)W[idx];
    }
}

// ---------------------------------------------------------------------------
__global__ void route_kernel(const float* __restrict__ x, int* __restrict__ sel_out,
                             int* __restrict__ sel_ws, int* __restrict__ counts) {
    const int i = blockIdx.x * blockDim.x + threadIdx.x;
    if (i >= B_N) return;
    const float xv = x[i * IN_F + 0];
    const float zv = x[i * IN_F + 2];
    const float two_pi = 6.28318530717958647692f;
    const float ang = atan2f(zv, xv);
    const float a = fmodf(ang + two_pi, two_pi) / two_pi * (float)NMODELS;
    int s = (int)floorf(a);
    s = s < 0 ? 0 : (s > NMODELS - 1 ? NMODELS - 1 : s);
    sel_out[i] = s;
    sel_ws[i] = s;
    atomicAdd(&counts[s], 1);
}

__global__ void fill_kernel(float4* __restrict__ logits, float4* __restrict__ probs) {
    const int i = blockIdx.x * blockDim.x + threadIdx.x;
    if (i < (B_N * NMODELS) / 4) {
        logits[i] = make_float4(1.0f, 1.0f, 1.0f, 1.0f);
        const float p = 1.0f / (float)NMODELS;
        probs[i] = make_float4(p, p, p, p);
    }
}

// Prefix sums + tile plan + workgroup plan (tiny; single thread).
__global__ void plan_kernel(const int* __restrict__ counts, int* __restrict__ cursor,
                            int* __restrict__ tileStart, int* __restrict__ tileCount,
                            int* __restrict__ wgExpert, int* __restrict__ wgFirst,
                            int* __restrict__ wgCnt, int* __restrict__ nWGs) {
    if (blockIdx.x == 0 && threadIdx.x == 0) {
        int off = 0, t = 0, w = 0;
        for (int e = 0; e < NMODELS; ++e) {
            cursor[e] = off;
            const int c = counts[e];
            const int tFirst = t;
            for (int j = 0; j < c; j += 16) {
                tileStart[t] = off + j;
                tileCount[t] = (c - j < 16) ? (c - j) : 16;
                ++t;
            }
            const int ntile = t - tFirst;
            for (int q = 0; q < ntile; q += 4) {
                wgExpert[w] = e;
                wgFirst[w] = tFirst + q;
                wgCnt[w] = (ntile - q < 4) ? (ntile - q) : 4;
                ++w;
            }
            off += c;
        }
        *nWGs = w;
    }
}

__global__ void scatter_kernel(const int* __restrict__ sel_ws, int* __restrict__ cursor,
                               int* __restrict__ perm) {
    const int i = blockIdx.x * blockDim.x + threadIdx.x;
    if (i >= B_N) return;
    const int rank = atomicAdd(&cursor[sel_ws[i]], 1);
    perm[rank] = i;
}

// ---------------------------------------------------------------------------
// 64 threads = 2 waves. One WG = up to 4 same-expert tiles; weights staged
// once into LDS (transposed f16), waves then run barrier-free with wave-local
// DS-counter fences. 60 WMMAs per tile.
// ---------------------------------------------------------------------------
__global__ void __launch_bounds__(64) moe_kernel(
    const float* __restrict__ x, const int* __restrict__ perm,
    const int* __restrict__ tileStart, const int* __restrict__ tileCount,
    const int* __restrict__ wgExpert, const int* __restrict__ wgFirst,
    const int* __restrict__ wgCnt, const int* __restrict__ nWGs,
    const float* __restrict__ W0, const float* __restrict__ b0,
    const float* __restrict__ W1, const float* __restrict__ b1,
    const float* __restrict__ W2, const float* __restrict__ b2,
    const float* __restrict__ W3, const float* __restrict__ b3,
    const float* __restrict__ W4, const float* __restrict__ b4,
    float* __restrict__ out) {
    const int wg = blockIdx.x;
    if (wg >= *nWGs) return;

    __shared__ __align__(32) _Float16 Wt0[64 * 32];            // 4 KB (K pad 32)
    __shared__ __align__(32) _Float16 Wt1[64 * 64];            // 8 KB
    __shared__ __align__(32) _Float16 Wt2s[64 * 64];
    __shared__ __align__(32) _Float16 Wt3s[64 * 64];
    __shared__ __align__(32) _Float16 Wt2e[64 * 64];
    __shared__ __align__(32) _Float16 Wt3e[64 * 64];
    __shared__ __align__(32) _Float16 Wt4s[16 * 64];           // 2 KB (N pad 16)
    __shared__ __align__(32) _Float16 Wt4e[16 * 64];
    __shared__ float b4sP[16], b4eP[16];
    __shared__ __align__(32) _Float16 act[NWAVES][3][16 * 64]; // 12 KB
    __shared__ int rowsArr[NWAVES][16];

    const int tid = threadIdx.x;
    const int e = wgExpert[wg];
    const float* W2e = W2 + (size_t)(e + 1) * HID * HID;
    const float* W3e = W3 + (size_t)(e + 1) * HID * HID;
    const float* W4e = W4 + (size_t)(e + 1) * HID * OUT_F;
    const float* b2e = b2 + (size_t)(e + 1) * HID;
    const float* b3e = b3 + (size_t)(e + 1) * HID;
    const float* b4e = b4 + (size_t)(e + 1) * OUT_F;

    // ---- weight staging (coalesced, unconditional global loads) ----
    stage_w64(W1, Wt1, tid);
    stage_w64(W2, Wt2s, tid);
    stage_w64(W3, Wt3s, tid);
    stage_w64(W2e, Wt2e, tid);
    stage_w64(W3e, Wt3e, tid);
    for (int idx = tid; idx < 64 * 32; idx += 64) Wt0[idx] = (_Float16)0.0f;
    for (int idx = tid; idx < 16 * 64; idx += 64) {
        Wt4s[idx] = (_Float16)0.0f;
        Wt4e[idx] = (_Float16)0.0f;
    }
    if (tid < 16) { b4sP[tid] = 0.0f; b4eP[tid] = 0.0f; }
    if (tid < OUT_F) { b4sP[tid] = b4[tid]; b4eP[tid] = b4e[tid]; }
    __syncthreads();
    for (int idx = tid; idx < IN_F * 64; idx += 64) {          // W0: 6x64
        const int k = idx >> 6, n = idx & 63;
        Wt0[n * 32 + k] = (_Float16)W0[idx];
    }
    for (int idx = tid; idx < HID * OUT_F; idx += 64) {        // W4: 64x3
        const int k = idx / OUT_F, n = idx - k * OUT_F;
        Wt4s[n * 64 + k] = (_Float16)W4[idx];
        Wt4e[n * 64 + k] = (_Float16)W4e[idx];
    }
    __syncthreads();   // weights ready; no cross-wave deps after this point

    // ---- per-wave tile processing ----
    const int wave = tid >> 5, lane = tid & 31;
    const int m = lane & 15, g = lane >> 4;
    _Float16* buf0 = act[wave][0];
    _Float16* buf1 = act[wave][1];
    _Float16* buf2 = act[wave][2];
    int* rows = rowsArr[wave];
    const int first = wgFirst[wg];
    const int nT = wgCnt[wg];

    for (int j = wave; j < nT; j += NWAVES) {
        const int tt = first + j;
        const int start = tileStart[tt];
        const int cnt = tileCount[tt];
        if (lane < 16) rows[lane] = perm[start + (lane < cnt ? lane : cnt - 1)];
        // zero the 16x32 x-staging region of buf0 (one 32B store per lane)
        v16h zv;
#pragma unroll
        for (int q = 0; q < 16; ++q) zv[q] = (_Float16)0.0f;
        *(v16h*)(buf0 + m * 64 + g * 16) = zv;
        wave_lds_fence();
        // gather 16 rows x 6 features, unconditional (96 = 3 * 32 lanes)
        for (int idx = lane; idx < 16 * IN_F; idx += 32) {
            const int r = idx / IN_F, col = idx - r * IN_F;
            buf0[r * 64 + col] = (_Float16)x[rows[r] * IN_F + col];
        }
        wave_lds_fence();

        dense_in(buf0, buf1, Wt0, b0, lane);                   // a1
        wave_lds_fence();
        dense64(buf1, buf0, Wt1, b1, lane);                    // a2
        wave_lds_fence();
        dense64(buf0, buf2, Wt2s, b2, lane);                   // a3s
        wave_lds_fence();
        dense64(buf2, buf1, Wt3s, b3, lane);                   // a4s
        wave_lds_fence();

        v8f c0 = head(buf1, Wt4s, b4sP, lane);                 // LOD0
        store_out(out, rows, cnt, 0, c0, lane);
        v8f c1 = head(buf1, Wt4e, b4eP, lane);                 // LOD1
        store_out(out, rows, cnt, 1, c1, lane);
        wave_lds_fence();                                      // WAR on buf1

        dense64(buf2, buf1, Wt3e, b3e, lane);                  // expert L3(a3s)
        wave_lds_fence();
        v8f c2 = head(buf1, Wt4e, b4eP, lane);                 // LOD2
        store_out(out, rows, cnt, 2, c2, lane);
        wave_lds_fence();                                      // WAR on buf2

        dense64(buf0, buf2, Wt2e, b2e, lane);                  // expert L2(a2)
        wave_lds_fence();
        dense64(buf2, buf0, Wt3e, b3e, lane);                  // expert L3
        wave_lds_fence();
        v8f c3 = head(buf0, Wt4e, b4eP, lane);                 // LOD3
        store_out(out, rows, cnt, 3, c3, lane);
        wave_lds_fence();                                      // before next tile
    }
}

// ---------------------------------------------------------------------------
extern "C" void kernel_launch(void* const* d_in, const int* in_sizes, int n_in,
                              void* d_out, int out_size, void* d_ws, size_t ws_size,
                              hipStream_t stream) {
    const float* x  = (const float*)d_in[0];
    const float* W0 = (const float*)d_in[1];
    const float* b0 = (const float*)d_in[2];
    const float* W1 = (const float*)d_in[3];
    const float* b1 = (const float*)d_in[4];
    const float* W2 = (const float*)d_in[5];
    const float* b2 = (const float*)d_in[6];
    const float* W3 = (const float*)d_in[7];
    const float* b3 = (const float*)d_in[8];
    const float* W4 = (const float*)d_in[9];
    const float* b4 = (const float*)d_in[10];

    // Output: model_outputs (B*4*3 f32) | sel_idx (B i32) |
    //         sel_logits (B*64 f32) | sel_probs (B*64 f32)
    float* out_model  = (float*)d_out;
    int*   out_sel    = (int*)(out_model + (size_t)B_N * 4 * OUT_F);
    float* out_logits = (float*)(out_sel + B_N);
    float* out_probs  = out_logits + (size_t)B_N * NMODELS;

    // Workspace (ints)
    int* counts     = (int*)d_ws;              // 64
    int* cursor     = counts + 64;             // 64
    int* nWGs       = cursor + 64;             // 1 (padded 64)
    int* tileStart  = nWGs + 64;               // MAX_TILES
    int* tileCount  = tileStart + MAX_TILES;   // MAX_TILES
    int* wgExpert   = tileCount + MAX_TILES;   // MAX_WGS
    int* wgFirst    = wgExpert + MAX_WGS;      // MAX_WGS
    int* wgCnt      = wgFirst + MAX_WGS;       // MAX_WGS
    int* sel_ws     = wgCnt + MAX_WGS;         // B
    int* perm       = sel_ws + B_N;            // B

    hipMemsetAsync(counts, 0, 64 * sizeof(int), stream);

    route_kernel<<<(B_N + 255) / 256, 256, 0, stream>>>(x, out_sel, sel_ws, counts);
    fill_kernel<<<((B_N * NMODELS) / 4 + 255) / 256, 256, 0, stream>>>(
        (float4*)out_logits, (float4*)out_probs);
    plan_kernel<<<1, 1, 0, stream>>>(counts, cursor, tileStart, tileCount,
                                     wgExpert, wgFirst, wgCnt, nWGs);
    scatter_kernel<<<(B_N + 255) / 256, 256, 0, stream>>>(sel_ws, cursor, perm);
    moe_kernel<<<MAX_WGS, 64, 0, stream>>>(x, perm, tileStart, tileCount,
                                           wgExpert, wgFirst, wgCnt, nWGs,
                                           W0, b0, W1, b1, W2, b2, W3, b3, W4, b4,
                                           out_model);
}
